// Level1Quantizer_80616536146013
// MI455X (gfx1250) — matrix operand order repaired and checked by reading, not compiled, not run on hardware
//
#include <hip/hip_runtime.h>
#include <hip/hip_bf16.h>

typedef __attribute__((ext_vector_type(16))) __bf16 v16bf;
typedef __attribute__((ext_vector_type(8)))  __bf16 v8bf;
typedef __attribute__((ext_vector_type(4)))  __bf16 v4bf;
typedef __attribute__((ext_vector_type(8)))  float  v8f;

#define DD   128
#define KCB  8
#define YSTR 136            // padded row stride (bf16 elems) -> conflict-free frag loads
#define ROWS_PER_BLOCK 128
#define NTHREADS 256
#define NWAVES 8

__global__ __launch_bounds__(NTHREADS)
void l1q_fused_kernel(const float* __restrict__ X,
                      const float* __restrict__ W,
                      const float* __restrict__ bias,
                      const float* __restrict__ gamma,
                      const float* __restrict__ beta,
                      const float* __restrict__ CB,
                      float* __restrict__ out_idx,
                      float* __restrict__ out_soft,
                      float* __restrict__ out_emb,
                      float* __restrict__ out_log)
{
    __shared__ __align__(16) __bf16 Wl[DD * DD];          // W bf16, row-major [n][k]
    __shared__ __align__(16) __bf16 cbnb[16 * YSTR];      // L2-normed codebook bf16, rows 8-15 zero
    __shared__ __align__(16) float  cbk[KCB * DD];        // raw codebook (embedding gather)
    __shared__ __align__(16) __bf16 yl[NWAVES * 16 * YSTR]; // per-wave LN output, row-major [m][k]
    __shared__ float rnl[NWAVES * 16];                    // per-row 1/||y||
    __shared__ int   idxl[NWAVES * 16];                   // per-row argmax

    const int t    = threadIdx.x;
    const int lane = t & 31;
    const int wv   = t >> 5;          // wave id 0..7
    const int half = lane >> 4;       // 0: lanes 0-15, 1: lanes 16-31
    const int l15  = lane & 15;

    // ---- stage W: fp32 -> bf16 LDS (16384 elems, 64/thread) ----
    const float4* W4 = (const float4*)W;
    #pragma unroll
    for (int p = 0; p < (DD * DD / 4) / NTHREADS; ++p) {
        int q = t + p * NTHREADS;
        float4 w = W4[q];
        v4bf b;
        b[0] = (__bf16)w.x; b[1] = (__bf16)w.y;
        b[2] = (__bf16)w.z; b[3] = (__bf16)w.w;
        *(v4bf*)&Wl[q * 4] = b;
    }
    // ---- stage raw codebook (1024 floats = 256 float4, one per thread) ----
    ((float4*)cbk)[t] = ((const float4*)CB)[t];
    // ---- normalized codebook (bf16): wave wv owns row wv; zero row 8+wv ----
    {
        float4 c = ((const float4*)CB)[wv * 32 + lane];
        float ss = c.x * c.x + c.y * c.y + c.z * c.z + c.w * c.w;
        #pragma unroll
        for (int m = 1; m < 32; m <<= 1) ss += __shfl_xor(ss, m, 32);
        float rnv = 1.0f / fmaxf(sqrtf(ss), 1e-12f);
        v4bf cn;
        cn[0] = (__bf16)(c.x * rnv); cn[1] = (__bf16)(c.y * rnv);
        cn[2] = (__bf16)(c.z * rnv); cn[3] = (__bf16)(c.w * rnv);
        *(v4bf*)&cbnb[wv * YSTR + lane * 4] = cn;
        v4bf z;
        z[0] = (__bf16)0.0f; z[1] = (__bf16)0.0f; z[2] = (__bf16)0.0f; z[3] = (__bf16)0.0f;
        *(v4bf*)&cbnb[(8 + wv) * YSTR + lane * 4] = z;
    }
    __syncthreads();

    // =============== GEMM: h = X * W^T  (16 rows per wave) ===============
    const long mBase = (long)blockIdx.x * ROWS_PER_BLOCK + wv * 16;
    const float* arow = X + (mBase + l15) * DD;
    const int kA = half * 8;    // A-frag K base offset within 32-chunk
    const int kB = half * 8;    // B-frag K base offset within 32-chunk

    v8f acc[8];
    #pragma unroll
    for (int nt = 0; nt < 8; ++nt)
        #pragma unroll
        for (int i = 0; i < 8; ++i) acc[nt][i] = 0.0f;

    #pragma unroll
    for (int k0 = 0; k0 < DD; k0 += 32) {
        const float* ap = arow + k0 + kA;
        float4 a0 = *(const float4*)(ap);
        float4 a1 = *(const float4*)(ap + 4);
        float4 a2 = *(const float4*)(ap + 16);
        float4 a3 = *(const float4*)(ap + 20);
        v16bf Af;
        Af[0]  = (__bf16)a0.x; Af[1]  = (__bf16)a0.y; Af[2]  = (__bf16)a0.z; Af[3]  = (__bf16)a0.w;
        Af[4]  = (__bf16)a1.x; Af[5]  = (__bf16)a1.y; Af[6]  = (__bf16)a1.z; Af[7]  = (__bf16)a1.w;
        Af[8]  = (__bf16)a2.x; Af[9]  = (__bf16)a2.y; Af[10] = (__bf16)a2.z; Af[11] = (__bf16)a2.w;
        Af[12] = (__bf16)a3.x; Af[13] = (__bf16)a3.y; Af[14] = (__bf16)a3.z; Af[15] = (__bf16)a3.w;
        #pragma unroll
        for (int nt = 0; nt < 8; ++nt) {
            const __bf16* bp = &Wl[(nt * 16 + l15) * DD + k0 + kB];
            v8bf blo = *(const v8bf*)bp;
            v8bf bhi = *(const v8bf*)(bp + 16);
            v16bf Bf = __builtin_shufflevector(blo, bhi,
                0, 1, 2, 3, 4, 5, 6, 7, 8, 9, 10, 11, 12, 13, 14, 15);
            acc[nt] = __builtin_amdgcn_wmma_f32_16x16x32_bf16(
                false, Af, false, Bf, (short)0, acc[nt], false, false);
        }
    }

    // =============== bias + LayerNorm ===============
    #pragma unroll
    for (int nt = 0; nt < 8; ++nt) {
        float bn = bias[nt * 16 + l15];
        #pragma unroll
        for (int i = 0; i < 8; ++i) acc[nt][i] += bn;
    }
    float mu[8], rs[8];
    #pragma unroll
    for (int i = 0; i < 8; ++i) {
        float s = 0.f, ss = 0.f;
        #pragma unroll
        for (int nt = 0; nt < 8; ++nt) { float v = acc[nt][i]; s += v; ss += v * v; }
        #pragma unroll
        for (int m = 1; m < 16; m <<= 1) { s += __shfl_xor(s, m, 32); ss += __shfl_xor(ss, m, 32); }
        float mean = s * (1.0f / DD);
        float var  = ss * (1.0f / DD) - mean * mean;
        mu[i] = mean;
        rs[i] = rsqrtf(var + 1e-5f);
    }
    #pragma unroll
    for (int nt = 0; nt < 8; ++nt) {
        float g  = gamma[nt * 16 + l15];
        float be = beta[nt * 16 + l15];
        #pragma unroll
        for (int i = 0; i < 8; ++i)
            acc[nt][i] = (acc[nt][i] - mu[i]) * rs[i] * g + be;
    }

    // =============== row L2 norm (folded into logits later) ===============
    #pragma unroll
    for (int i = 0; i < 8; ++i) {
        float ss = 0.f;
        #pragma unroll
        for (int nt = 0; nt < 8; ++nt) { float v = acc[nt][i]; ss += v * v; }
        #pragma unroll
        for (int m = 1; m < 16; m <<= 1) ss += __shfl_xor(ss, m, 32);
        float rnv = 1.0f / fmaxf(sqrtf(ss), 1e-12f);
        if (l15 == 0) rnl[wv * 16 + half * 8 + i] = rnv;  // uniform within half
    }

    // =============== spill y (bf16, row-major, padded) to LDS ===============
    __bf16* ylw = &yl[wv * 16 * YSTR];
    #pragma unroll
    for (int nt = 0; nt < 8; ++nt) {
        #pragma unroll
        for (int i = 0; i < 8; ++i)
            ylw[(i + half * 8) * YSTR + nt * 16 + l15] = (__bf16)acc[nt][i];
    }

    // =============== logits^T = cbn (A) x y^T (B) : one 16x16 tile ===============
    v8f lacc;
    #pragma unroll
    for (int j = 0; j < 8; ++j) lacc[j] = 0.0f;
    #pragma unroll
    for (int k0 = 0; k0 < DD; k0 += 32) {
        const __bf16* apc = &cbnb[l15 * YSTR + k0 + kA];
        v8bf a0 = *(const v8bf*)apc;
        v8bf a1 = *(const v8bf*)(apc + 16);
        v16bf Afc = __builtin_shufflevector(a0, a1,
            0, 1, 2, 3, 4, 5, 6, 7, 8, 9, 10, 11, 12, 13, 14, 15);
        const __bf16* bpy = &ylw[l15 * YSTR + k0 + kB];
        v8bf b0 = *(const v8bf*)bpy;
        v8bf b1 = *(const v8bf*)(bpy + 16);
        v16bf Bfy = __builtin_shufflevector(b0, b1,
            0, 1, 2, 3, 4, 5, 6, 7, 8, 9, 10, 11, 12, 13, 14, 15);
        lacc = __builtin_amdgcn_wmma_f32_16x16x32_bf16(
            false, Afc, false, Bfy, (short)0, lacc, false, false);
    }
    // C tile: VGPR j holds logits[j][m=l15] on lanes 0-15 (lanes 16-31: zero rows)

    // =============== per-lane softmax / argmax (lane = row, no shuffles) ===============
    float rnm = rnl[wv * 16 + l15];
    float lgv[8];
    float mx = -3.0e38f;
    #pragma unroll
    for (int j = 0; j < 8; ++j) { lgv[j] = lacc[j] * rnm; mx = fmaxf(mx, lgv[j]); }
    float es[8];
    float ssum = 0.f;
    #pragma unroll
    for (int j = 0; j < 8; ++j) { es[j] = __expf(lgv[j] - mx); ssum += es[j]; }
    float inv = 1.0f / ssum;
    int   bi = 0;
    float bv = lgv[0];
    #pragma unroll
    for (int j = 1; j < 8; ++j) {       // strict '>' keeps first max index
        bool take = lgv[j] > bv;
        bv = take ? lgv[j] : bv;
        bi = take ? j : bi;
    }

    if (half == 0) {                     // lanes 0-15 each own one row
        long r = mBase + l15;
        out_idx[r] = (float)bi;
        idxl[wv * 16 + l15] = bi;
        float4 s0, s1, g0, g1;
        s0.x = es[0] * inv; s0.y = es[1] * inv; s0.z = es[2] * inv; s0.w = es[3] * inv;
        s1.x = es[4] * inv; s1.y = es[5] * inv; s1.z = es[6] * inv; s1.w = es[7] * inv;
        g0.x = lgv[0]; g0.y = lgv[1]; g0.z = lgv[2]; g0.w = lgv[3];
        g1.x = lgv[4]; g1.y = lgv[5]; g1.z = lgv[6]; g1.w = lgv[7];
        *(float4*)&out_soft[r * KCB]     = s0;
        *(float4*)&out_soft[r * KCB + 4] = s1;
        *(float4*)&out_log[r * KCB]      = g0;
        *(float4*)&out_log[r * KCB + 4]  = g1;
    }

    // =============== embedding = raw codebook row, cooperative copy ===============
    #pragma unroll
    for (int i = 0; i < 8; ++i) {
        int row = i + half * 8;
        int bidx = idxl[wv * 16 + row];          // same-wave LDS, DS ordering suffices
        long r = mBase + row;
        float4 e0 = *(const float4*)&cbk[bidx * DD + l15 * 4];
        float4 e1 = *(const float4*)&cbk[bidx * DD + 64 + l15 * 4];
        *(float4*)&out_emb[r * DD + l15 * 4]      = e0;
        *(float4*)&out_emb[r * DD + 64 + l15 * 4] = e1;
    }
}

extern "C" void kernel_launch(void* const* d_in, const int* in_sizes, int n_in,
                              void* d_out, int out_size, void* d_ws, size_t ws_size,
                              hipStream_t stream)
{
    (void)n_in; (void)out_size; (void)d_ws; (void)ws_size;
    const float* X     = (const float*)d_in[0];
    const float* W     = (const float*)d_in[1];
    const float* bias  = (const float*)d_in[2];
    const float* gamma = (const float*)d_in[3];
    const float* beta  = (const float*)d_in[4];
    const float* CB    = (const float*)d_in[5];
    const long B = (long)in_sizes[0] / DD;   // 262144

    float* out      = (float*)d_out;
    float* out_idx  = out;                        // (B,)
    float* out_soft = out + B;                    // (B, 8)
    float* out_emb  = out + B + B * KCB;          // (B, 128)
    float* out_log  = out + B + B * KCB + B * DD; // (B, 8)

    dim3 grid((unsigned)(B / ROWS_PER_BLOCK));
    l1q_fused_kernel<<<grid, NTHREADS, 0, stream>>>(
        X, W, bias, gamma, beta, CB, out_idx, out_soft, out_emb, out_log);
}
